// SE3Transformer_90640989815136
// MI455X (gfx1250) — compile-verified, matrix operand order checked by reading, not run on hardware
//
#include <hip/hip_runtime.h>
#include <hip/hip_fp16.h>
#include <string.h>
#include <math.h>

typedef __attribute__((ext_vector_type(16))) _Float16 v16h;
typedef __attribute__((ext_vector_type(8)))  float    v8f;

static constexpr int NN  = 2048;    // nodes
static constexpr int NE  = 32768;   // edges
static constexpr int ND  = 4;       // degrees
static constexpr int NC  = 32;      // channels
static constexpr int NH  = 4;       // heads
static constexpr int MC  = 8;       // mid channels
static constexpr int RD  = 128;     // radial hidden dim
static constexpr int AFS = 64;      // atom feature size
static constexpr int AE  = 256;     // atom embedding
static constexpr int ECH = 4096;    // edge chunk for W3 / message fusion
static constexpr int EFP = 32;      // padded edge-feature width (logical 5)

__host__ __device__ inline int imin2(int a, int b) { return a < b ? a : b; }

__host__ __device__ inline size_t basis_pair_base(int DI, int DO) {
  size_t off = 0;
  for (int di = 0; di < ND; di++)
    for (int d0 = 0; d0 < ND; d0++) {
      if (di == DI && d0 == DO) return off;
      int F = 2 * imin2(di, d0) + 1;
      off += (size_t)NE * F * (2 * d0 + 1) * (2 * di + 1);
    }
  return off;
}
__host__ __device__ inline size_t basis_total() {
  size_t off = 0;
  for (int di = 0; di < ND; di++)
    for (int d0 = 0; d0 < ND; d0++) {
      int F = 2 * imin2(di, d0) + 1;
      off += (size_t)NE * F * (2 * d0 + 1) * (2 * di + 1);
    }
  return off;
}
__host__ __device__ inline int q_base(int DI, int DO, int JJ) {
  int off = 0;
  for (int di = 0; di < ND; di++)
    for (int d0 = 0; d0 < ND; d0++)
      for (int J = (di > d0 ? di - d0 : d0 - di); J <= di + d0; J++) {
        if (di == DI && d0 == DO && J == JJ) return off;
        off += (2 * d0 + 1) * (2 * di + 1) * (2 * J + 1);
      }
  return off;
}

// ---------------------------------------------------------------------------
// Clebsch-Gordan / Q-matrix computation (device, double precision, per call)
// ---------------------------------------------------------------------------
struct cpx { double x, y; };

__device__ inline double factd(int n) { double r = 1.0; for (int i = 2; i <= n; i++) r *= i; return r; }

__device__ double cgc(int j1, int m1, int j2, int m2, int j3, int m3) {
  if (m1 + m2 != m3) return 0.0;
  double pre = sqrt((2.0 * j3 + 1.0) * factd(j3 + j1 - j2) * factd(j3 - j1 + j2) *
                    factd(j1 + j2 - j3) / factd(j1 + j2 + j3 + 1));
  pre *= sqrt(factd(j3 + m3) * factd(j3 - m3) * factd(j1 - m1) * factd(j1 + m1) *
              factd(j2 - m2) * factd(j2 + m2));
  int kmin = 0;
  if (j2 - j3 - m1 > kmin) kmin = j2 - j3 - m1;
  if (j1 - j3 + m2 > kmin) kmin = j1 - j3 + m2;
  int kmax = j1 + j2 - j3;
  if (j1 - m1 < kmax) kmax = j1 - m1;
  if (j2 + m2 < kmax) kmax = j2 + m2;
  double s = 0.0;
  for (int k = kmin; k <= kmax; k++) {
    double d = factd(k) * factd(j1 + j2 - j3 - k) * factd(j1 - m1 - k) *
               factd(j2 + m2 - k) * factd(j3 - j2 + m1 + k) * factd(j3 - j1 - m2 + k);
    s += ((k & 1) ? -1.0 : 1.0) / d;
  }
  return pre * s;
}

__device__ inline cpx realC(int l, int x, int a) {
  int r = x - l, c = a - l;
  const double s2 = 0.70710678118654752440;
  cpx z = {0.0, 0.0};
  if (r == 0 && c == 0) { z.x = 1.0; return z; }
  int m = (r < 0) ? -r : r;
  double sgn = (m & 1) ? -1.0 : 1.0;
  if (r > 0 && c ==  r) { z.x = sgn * s2; return z; }
  if (r > 0 && c == -r) { z.x = s2;       return z; }
  if (r < 0 && c ==  r) { z.y = s2;       return z; }
  if (r < 0 && c == -r) { z.y = -sgn * s2; return z; }
  return z;
}

__global__ __launch_bounds__(256) void compute_q_kernel(float* __restrict__ Qbuf) {
  __shared__ double CG[637];
  __shared__ double Mre[637];
  __shared__ double Mim[637];
  __shared__ double redR[256];
  __shared__ double redI[256];
  int tid = threadIdx.x;
  int target = blockIdx.x;
  int DI = 0, DO = 0, JJ = 0, t = 0;
  for (int di = 0; di < ND; di++)
    for (int d0 = 0; d0 < ND; d0++)
      for (int J = (di > d0 ? di - d0 : d0 - di); J <= di + d0; J++) {
        if (t == target) { DI = di; DO = d0; JJ = J; }
        t++;
      }
  int na = 2 * DO + 1, nb = 2 * DI + 1, nc = 2 * JJ + 1, tot = na * nb * nc;
  for (int idx = tid; idx < tot; idx += 256) {
    int a = idx / (nb * nc), b = (idx / nc) % nb, c = idx % nc;
    CG[idx] = cgc(DO, a - DO, DI, b - DI, JJ, c - JJ);
  }
  __syncthreads();
  for (int idx = tid; idx < tot; idx += 256) {
    int x = idx / (nb * nc), y = (idx / nc) % nb, z = idx % nc;
    double re = 0.0, im = 0.0;
    for (int a = 0; a < na; a++) {
      cpx ca = realC(DO, x, a);
      if (ca.x == 0.0 && ca.y == 0.0) continue;
      for (int b = 0; b < nb; b++) {
        cpx cb = realC(DI, y, b);
        if (cb.x == 0.0 && cb.y == 0.0) continue;
        double t1r = ca.x * cb.x - ca.y * cb.y;
        double t1i = ca.x * cb.y + ca.y * cb.x;
        for (int c = 0; c < nc; c++) {
          double g = CG[(a * nb + b) * nc + c];
          if (g == 0.0) continue;
          cpx cc = realC(JJ, z, c);           // multiply by conj(cc)
          re += (t1r * cc.x + t1i * cc.y) * g;
          im += (t1i * cc.x - t1r * cc.y) * g;
        }
      }
    }
    Mre[idx] = re; Mim[idx] = im;
  }
  __syncthreads();
  double sr = 0.0, si = 0.0;
  for (int idx = tid; idx < tot; idx += 256) { sr += fabs(Mre[idx]); si += fabs(Mim[idx]); }
  redR[tid] = sr; redI[tid] = si;
  __syncthreads();
  for (int s = 128; s > 0; s >>= 1) {
    if (tid < s) { redR[tid] += redR[tid + s]; redI[tid] += redI[tid + s]; }
    __syncthreads();
  }
  bool useRe = redR[0] >= redI[0];
  int base = q_base(DI, DO, JJ);
  for (int idx = tid; idx < tot; idx += 256)
    Qbuf[base + idx] = (float)(useRe ? Mre[idx] : Mim[idx]);
}

// ---------------------------------------------------------------------------
// Spherical harmonics + equivariant basis (one thread per edge)
// ---------------------------------------------------------------------------
__device__ inline float factf(int n) { float r = 1.f; for (int i = 2; i <= n; i++) r *= (float)i; return r; }

__device__ void sph_all(float px, float py, float pz, float* Y) {
  float r  = sqrtf(px * px + py * py + pz * pz + 1e-12f);
  float ct = pz / r;
  float st = sqrtf(fmaxf(1.f - ct * ct, 1e-12f));
  float phi = atan2f(py, px);
  float P[7][7];
  P[0][0] = 1.f;
  for (int m = 1; m <= 6; m++) P[m][m] = -(2.f * m - 1.f) * st * P[m - 1][m - 1];
  for (int m = 0; m < 6; m++)  P[m + 1][m] = (2.f * m + 1.f) * ct * P[m][m];
  for (int m = 0; m <= 6; m++)
    for (int l = m + 2; l <= 6; l++)
      P[l][m] = ((2.f * l - 1.f) * ct * P[l - 1][m] - (float)(l + m - 1) * P[l - 2][m]) / (float)(l - m);
  const float inv4pi = 0.07957747154594767f;
  const float s2 = 1.4142135623730951f;
  for (int l = 0; l <= 6; l++) {
    for (int m = -l; m <= l; m++) {
      int am = m < 0 ? -m : m;
      float K = sqrtf((2.f * l + 1.f) * inv4pi * factf(l - am) / factf(l + am));
      float v;
      if (m == 0)      v = K * P[l][0];
      else if (m > 0)  v = s2 * K * cosf((float)m * phi) * P[l][m];
      else             v = s2 * K * sinf((float)am * phi) * P[l][am];
      Y[l * l + (m + l)] = v;
    }
  }
}

__global__ void basis_kernel(const float* __restrict__ rel, const float* __restrict__ Qbuf,
                             float* __restrict__ basisbuf) {
  int e = blockIdx.x * blockDim.x + threadIdx.x;
  if (e >= NE) return;
  float Y[49];
  sph_all(rel[e * 3 + 0], rel[e * 3 + 1], rel[e * 3 + 2], Y);
  for (int di = 0; di < ND; di++)
    for (int d0 = 0; d0 < ND; d0++) {
      int ao = 2 * d0 + 1, bo = 2 * di + 1;
      int F = 2 * imin2(di, d0) + 1;
      size_t pb = basis_pair_base(di, d0);
      int f = 0;
      for (int J = (di > d0 ? di - d0 : d0 - di); J <= di + d0; J++, f++) {
        int nc = 2 * J + 1;
        int qb = q_base(di, d0, J);
        for (int a = 0; a < ao; a++)
          for (int b = 0; b < bo; b++) {
            float s = 0.f;
            for (int c = 0; c < nc; c++) s += Y[J * J + c] * Qbuf[qb + (a * bo + b) * nc + c];
            basisbuf[pb + ((size_t)e * F + f) * (size_t)(ao * bo) + a * bo + b] = s;
          }
      }
    }
}

// writes zero-padded edge features, row stride EFP (=32) so all GEMM K%32==0
__global__ void edgefeat_kernel(const float* __restrict__ rel, const float* __restrict__ ew,
                                float* __restrict__ ef) {
  int e = blockIdx.x * blockDim.x + threadIdx.x;
  if (e >= NE) return;
  float x = rel[e * 3 + 0], y = rel[e * 3 + 1], z = rel[e * 3 + 2];
  float* row = ef + (size_t)e * EFP;
  row[0] = sqrtf(x * x + y * y + z * z);
  row[1] = ew[e * 4 + 0];
  row[2] = ew[e * 4 + 1];
  row[3] = ew[e * 4 + 2];
  row[4] = ew[e * 4 + 3];
  for (int j = 5; j < EFP; j++) row[j] = 0.f;
}

// ---------------------------------------------------------------------------
// Weight pre-swizzle: B[K,N] (f32, zero-padded to Kpad rows) -> f16 fragments.
// Layout: tile (kc, ct) occupies 512 halves; lane L owns halves [L*16, L*16+16)
// with element j <-> B[kc*32 + kk(j,L)][ct*16 + (L&15)],
// kk = (j<8 ? j : j+8) + (L>=16 ? 8 : 0).
// ---------------------------------------------------------------------------
__global__ void wswz_kernel(const float* __restrict__ W, _Float16* __restrict__ out,
                            int K, int N, int Kpad) {
  size_t tot = (size_t)(Kpad >> 5) * (N >> 4) * 512;
  size_t st = (size_t)gridDim.x * blockDim.x;
  int ntiles = N >> 4;
  for (size_t i = (size_t)blockIdx.x * blockDim.x + threadIdx.x; i < tot; i += st) {
    int within = (int)(i & 511);
    size_t tile = i >> 9;
    int lane = within >> 4, j = within & 15;
    int kc = (int)(tile / ntiles), ct = (int)(tile % ntiles);
    int lo = lane & 15, hi = lane >> 4;
    int kk = (j < 8 ? j : j + 8) + (hi ? 8 : 0);
    int k = kc * 32 + kk, col = ct * 16 + lo;
    float v = (k < K) ? W[(size_t)k * N + col] : 0.f;
    out[i] = (_Float16)v;
  }
}

// ---------------------------------------------------------------------------
// WMMA GEMM: C[M,N] = act(A[M,K]*B + bias).  Requires M%128==0, N%16==0,
// Kpad%32==0 (guaranteed by construction).  B pre-swizzled f16 (wswz_kernel).
// 8 waves / block, one 16x16 C tile per wave.  Inner loop: 4x b128 A loads,
// 1x 32B B fragment load, cvt, v_wmma.
// ---------------------------------------------------------------------------
__global__ __launch_bounds__(256) void gemm_wmma_kernel(
    const float* __restrict__ A, const _Float16* __restrict__ Bh,
    const float* __restrict__ bias, float* __restrict__ C,
    int M, int N, int Kpad, int lda, int ldc, int relu) {
  int wave = threadIdx.x >> 5;
  int lane = threadIdx.x & 31;
  int col0 = blockIdx.x * 16;
  int row0 = blockIdx.y * 128 + wave * 16;
  int lo = lane & 15;
  int hi = lane >> 4;
  int ntiles = N >> 4;
  v8f acc = {};
  const float* arow = A + (size_t)(row0 + lo) * lda + hi * 8;
  const v16h* bbase = (const v16h*)Bh;
  for (int k0 = 0; k0 < Kpad; k0 += 32) {
    const float4* a0 = (const float4*)(arow + k0);
    const float4* a1 = (const float4*)(arow + k0 + 16);
    float4 x0 = a0[0], x1 = a0[1];
    float4 y0 = a1[0], y1 = a1[1];
    if (k0 + 32 < Kpad) __builtin_prefetch(arow + k0 + 32, 0, 0);
    v16h bf = bbase[((size_t)(k0 >> 5) * ntiles + (col0 >> 4)) * 32 + lane];
    v16h af;
    af[0]  = (_Float16)x0.x; af[1]  = (_Float16)x0.y;
    af[2]  = (_Float16)x0.z; af[3]  = (_Float16)x0.w;
    af[4]  = (_Float16)x1.x; af[5]  = (_Float16)x1.y;
    af[6]  = (_Float16)x1.z; af[7]  = (_Float16)x1.w;
    af[8]  = (_Float16)y0.x; af[9]  = (_Float16)y0.y;
    af[10] = (_Float16)y0.z; af[11] = (_Float16)y0.w;
    af[12] = (_Float16)y1.x; af[13] = (_Float16)y1.y;
    af[14] = (_Float16)y1.z; af[15] = (_Float16)y1.w;
    acc = __builtin_amdgcn_wmma_f32_16x16x32_f16(false, af, false, bf, (short)0, acc, false, false);
  }
  int col = col0 + lo;
  float bv = bias ? bias[col] : 0.f;
#pragma unroll
  for (int r = 0; r < 8; r++) {
    int row = row0 + r + hi * 8;
    float v = acc[r] + bv;
    if (relu) v = fmaxf(v, 0.f);
    C[(size_t)row * ldc + col] = v;
  }
}

// ---------------------------------------------------------------------------
// BatchNorm (over edge batch) + relu
// ---------------------------------------------------------------------------
__global__ __launch_bounds__(256) void bn_stats_kernel(const float* __restrict__ X, int Mrows, int ld,
                                                       float* __restrict__ mu, float* __restrict__ rstd) {
  __shared__ float s1[256];
  __shared__ float s2[256];
  int c = blockIdx.x, tid = threadIdx.x;
  float a = 0.f, b = 0.f;
  for (int r = tid; r < Mrows; r += 256) {
    float v = X[(size_t)r * ld + c];
    a += v; b += v * v;
  }
  s1[tid] = a; s2[tid] = b;
  __syncthreads();
  for (int s = 128; s > 0; s >>= 1) {
    if (tid < s) { s1[tid] += s1[tid + s]; s2[tid] += s2[tid + s]; }
    __syncthreads();
  }
  if (tid == 0) {
    float m = s1[0] / Mrows;
    float v = s2[0] / Mrows - m * m;
    mu[c] = m;
    rstd[c] = rsqrtf(v + 1e-5f);
  }
}

__global__ void bn_apply_relu_kernel(float* __restrict__ X, int Mrows, int ld,
                                     const float* __restrict__ mu, const float* __restrict__ rstd,
                                     const float* __restrict__ g, const float* __restrict__ be) {
  size_t tot = (size_t)Mrows * ld;
  size_t st = (size_t)gridDim.x * blockDim.x;
  for (size_t i = (size_t)blockIdx.x * blockDim.x + threadIdx.x; i < tot; i += st) {
    int c = (int)(i % ld);
    float v = (X[i] - mu[c]) * rstd[c] * g[c] + be[c];
    X[i] = fmaxf(v, 0.f);
  }
}

// ---------------------------------------------------------------------------
// Utility fills
// ---------------------------------------------------------------------------
__global__ void fill_kernel(float* p, size_t n, float v) {
  size_t st = (size_t)gridDim.x * blockDim.x;
  for (size_t i = (size_t)blockIdx.x * blockDim.x + threadIdx.x; i < n; i += st) p[i] = v;
}
__global__ void fillu_kernel(unsigned* p, size_t n, unsigned v) {
  size_t st = (size_t)gridDim.x * blockDim.x;
  for (size_t i = (size_t)blockIdx.x * blockDim.x + threadIdx.x; i < n; i += st) p[i] = v;
}

// ---------------------------------------------------------------------------
// Per-edge pair message:  msg[e,o,a] += sum_{f,i} R[e,f,o,i] *
//                         (sum_b B[e,f,a,b] * feat[src[e],i,b])
// One wave per edge; i (m_in) striped over the 32 lanes; per-edge basis in LDS.
// ---------------------------------------------------------------------------
__global__ __launch_bounds__(256) void pair_msg_kernel(
    const float* __restrict__ Bp, const float* __restrict__ R, const float* __restrict__ feat,
    const int* __restrict__ src, float* __restrict__ msg,
    int e0, int e1, int F, int ao, int bo, int m_in, int m_out) {
  __shared__ float Bs[8][352];
  int wave = threadIdx.x >> 5, lane = threadIdx.x & 31;
  int e = e0 + blockIdx.x * 8 + wave;
  bool valid = e < e1;
  int Fab = F * ao * bo;
  int eload = valid ? e : (e1 - 1);
  for (int idx = lane; idx < Fab; idx += 32) Bs[wave][idx] = Bp[(size_t)eload * Fab + idx];
  __syncthreads();
  if (!valid) return;
  float acc[56];
  int nacc = m_out * ao;
  for (int i = 0; i < nacc; i++) acc[i] = 0.f;
  int s = src[e];
  const float* fp = feat + (size_t)s * m_in * bo;
  const float* rp = R + (size_t)(e - e0) * F * m_out * m_in;
  for (int t = 0; t < m_in; t += 32) {
    int i = t + lane;
    float fv[7];
    for (int b = 0; b < bo; b++) fv[b] = fp[(size_t)i * bo + b];
    for (int f = 0; f < F; f++) {
      float Ta[7];
      for (int a = 0; a < ao; a++) {
        float sv = 0.f;
        for (int b = 0; b < bo; b++) sv += Bs[wave][(f * ao + a) * bo + b] * fv[b];
        Ta[a] = sv;
      }
      const float* rfo = rp + (size_t)f * m_out * m_in + i;
      for (int o = 0; o < m_out; o++) {
        float rv = rfo[(size_t)o * m_in];
        for (int a = 0; a < ao; a++) acc[o * ao + a] += rv * Ta[a];
      }
    }
  }
  for (int i = 0; i < nacc; i++) {
    float v = acc[i];
    for (int off = 16; off > 0; off >>= 1) v += __shfl_xor(v, off, 32);
    acc[i] = v;
  }
  if (lane == 0) {
    float* mp = msg + (size_t)e * m_out * ao;
    for (int i = 0; i < nacc; i++) mp[i] += acc[i];
  }
}

// Degree-0-output message (reg/cls heads): m_out=128, F=1, ao=1, m_in=32.
__global__ __launch_bounds__(256) void pair_msg_deg0_kernel(
    const float* __restrict__ Bp, const float* __restrict__ R, const float* __restrict__ feat,
    const int* __restrict__ src, float* __restrict__ msg, int e0, int e1, int bo) {
  int wave = threadIdx.x >> 5, lane = threadIdx.x & 31;
  int e = e0 + blockIdx.x * 8 + wave;
  if (e >= e1) return;
  int s = src[e];
  float Ti = 0.f;
  {
    const float* fp = feat + (size_t)s * 32 * bo + (size_t)lane * bo;
    for (int b = 0; b < bo; b++) Ti += Bp[(size_t)e * bo + b] * fp[b];
  }
  const float* rp = R + (size_t)(e - e0) * 128 * 32;
  float o0 = 0.f, o1 = 0.f, o2 = 0.f, o3 = 0.f;
  for (int i = 0; i < 32; i++) {
    float ti = __shfl(Ti, i, 32);
    o0 += rp[(size_t)(lane      ) * 32 + i] * ti;
    o1 += rp[(size_t)(lane + 32 ) * 32 + i] * ti;
    o2 += rp[(size_t)(lane + 64 ) * 32 + i] * ti;
    o3 += rp[(size_t)(lane + 96 ) * 32 + i] * ti;
  }
  float* mp = msg + (size_t)e * 128;
  mp[lane] += o0; mp[lane + 32] += o1; mp[lane + 64] += o2; mp[lane + 96] += o3;
}

// ---------------------------------------------------------------------------
// 1x1 "graph conv": Y[n,o,m] = sum_c W[o,c] * concat(X1,X2)[n,c,m]
// ---------------------------------------------------------------------------
__global__ void g1x1_concat_kernel(const float* __restrict__ W, const float* __restrict__ X1, int c1,
                                   const float* __restrict__ X2, int c2, float* __restrict__ Y,
                                   int Nn, int oc, int m) {
  size_t tot = (size_t)Nn * oc * m;
  size_t st = (size_t)gridDim.x * blockDim.x;
  for (size_t idx = (size_t)blockIdx.x * blockDim.x + threadIdx.x; idx < tot; idx += st) {
    int mm = (int)(idx % m);
    int o  = (int)((idx / m) % oc);
    int n  = (int)(idx / ((size_t)m * oc));
    const float* wr = W + (size_t)o * (c1 + c2);
    float s = 0.f;
    const float* x1 = X1 + (size_t)n * c1 * m + mm;
    for (int c = 0; c < c1; c++) s += wr[c] * x1[(size_t)c * m];
    if (X2) {
      const float* x2 = X2 + (size_t)n * c2 * m + mm;
      for (int c = 0; c < c2; c++) s += wr[c1 + c] * x2[(size_t)c * m];
    }
    Y[idx] = s;
  }
}

// ---------------------------------------------------------------------------
// Attention: logits, segment softmax via atomics, weighted aggregation
// ---------------------------------------------------------------------------
__global__ void logits_kernel(const float* k0, const float* k1, const float* k2, const float* k3,
                              const float* q0, const float* q1, const float* q2, const float* q3,
                              const int* __restrict__ dst, float* __restrict__ lg,
                              int ndeg, float scale) {
  int e = blockIdx.x * blockDim.x + threadIdx.x;
  if (e >= NE) return;
  const float* ks[4] = {k0, k1, k2, k3};
  const float* qs[4] = {q0, q1, q2, q3};
  int dn = dst[e];
  float acc[4] = {0.f, 0.f, 0.f, 0.f};
  for (int d = 0; d < ndeg; d++) {
    int m = 2 * d + 1;
    const float* kv = ks[d] + (size_t)e  * MC * m;
    const float* qv = qs[d] + (size_t)dn * MC * m;
    for (int c = 0; c < MC; c++) {
      int h = c >> 1;
      for (int mm = 0; mm < m; mm++) acc[h] += kv[c * m + mm] * qv[c * m + mm];
    }
  }
  for (int h = 0; h < NH; h++) lg[e * NH + h] = acc[h] * scale;
}

__device__ inline unsigned encf(float f) {
  unsigned u = __float_as_uint(f);
  return (u & 0x80000000u) ? ~u : (u | 0x80000000u);
}
__device__ inline float decf(unsigned u) {
  unsigned b = (u & 0x80000000u) ? (u & 0x7fffffffu) : ~u;
  return __uint_as_float(b);
}

__global__ void segmax_kernel(const float* __restrict__ lg, const int* __restrict__ dst,
                              unsigned* __restrict__ mx) {
  int i = blockIdx.x * blockDim.x + threadIdx.x;
  if (i >= NE * NH) return;
  int e = i >> 2, h = i & 3;
  atomicMax(&mx[dst[e] * NH + h], encf(lg[i]));
}
__global__ void expden_kernel(const float* __restrict__ lg, const int* __restrict__ dst,
                              const unsigned* __restrict__ mx, float* __restrict__ ex,
                              float* __restrict__ den) {
  int i = blockIdx.x * blockDim.x + threadIdx.x;
  if (i >= NE * NH) return;
  int e = i >> 2, h = i & 3;
  float m = decf(mx[dst[e] * NH + h]);
  float v = expf(lg[i] - m);
  ex[i] = v;
  atomicAdd(&den[dst[e] * NH + h], v);
}
__global__ void alpha_kernel(const float* __restrict__ ex, const int* __restrict__ dst,
                             const float* __restrict__ den, float* __restrict__ al) {
  int i = blockIdx.x * blockDim.x + threadIdx.x;
  if (i >= NE * NH) return;
  int e = i >> 2, h = i & 3;
  al[i] = ex[i] / (den[dst[e] * NH + h] + 1e-9f);
}
__global__ void vagg_kernel(const float* __restrict__ vm, const float* __restrict__ alpha,
                            const int* __restrict__ dst, float* __restrict__ agg, int m) {
  size_t tot = (size_t)NE * MC * m;
  size_t st = (size_t)gridDim.x * blockDim.x;
  for (size_t i = (size_t)blockIdx.x * blockDim.x + threadIdx.x; i < tot; i += st) {
    int rem = (int)(i % (MC * m));
    int e   = (int)(i / (MC * m));
    int c   = rem / m;
    atomicAdd(&agg[(size_t)dst[e] * MC * m + rem], alpha[e * NH + (c >> 1)] * vm[i]);
  }
}

// ---------------------------------------------------------------------------
// Equivariant norm nonlinearity
// ---------------------------------------------------------------------------
__global__ void norm_nrm_kernel(const float* __restrict__ X, float* __restrict__ nrm, int m) {
  int i = blockIdx.x * blockDim.x + threadIdx.x;
  if (i >= NN * NC) return;
  const float* xp = X + (size_t)i * m;
  float s = 0.f;
  for (int mm = 0; mm < m; mm++) { float v = xp[mm]; s += v * v; }
  nrm[i] = sqrtf(s + 1e-12f);
}
__global__ void norm_apply_kernel(float* __restrict__ X, const float* __restrict__ nrm,
                                  const float* __restrict__ W, const float* __restrict__ b, int m) {
  int i = blockIdx.x * blockDim.x + threadIdx.x;
  if (i >= NN * NC) return;
  int n = i / NC, c = i % NC;
  float t = b[c];
  const float* nr = nrm + (size_t)n * NC;
  for (int c2 = 0; c2 < NC; c2++) t += nr[c2] * W[c2 * NC + c];
  t = fmaxf(t, 0.f);
  float inv = t / nrm[i];
  float* xp = X + (size_t)i * m;
  for (int mm = 0; mm < m; mm++) xp[mm] *= inv;
}

// ---------------------------------------------------------------------------
// Final graph convolution (reg/cls heads) + max pool
// ---------------------------------------------------------------------------
__global__ void degcount_kernel(const int* __restrict__ dst, float* __restrict__ deg) {
  int e = blockIdx.x * blockDim.x + threadIdx.x;
  if (e >= NE) return;
  atomicAdd(&deg[dst[e]], 1.f);
}
__global__ void eagg_kernel(const float* __restrict__ eb, const int* __restrict__ dst,
                            float* __restrict__ nodeb) {
  size_t tot = (size_t)NE * 128;
  size_t st = (size_t)gridDim.x * blockDim.x;
  for (size_t i = (size_t)blockIdx.x * blockDim.x + threadIdx.x; i < tot; i += st) {
    int e = (int)(i >> 7), o = (int)(i & 127);
    atomicAdd(&nodeb[(size_t)dst[e] * 128 + o], eb[i]);
  }
}
__global__ void conv_final_kernel(float* __restrict__ nodeb, const float* __restrict__ deg,
                                  const float* __restrict__ selfW, const float* __restrict__ h0d) {
  int i = blockIdx.x * blockDim.x + threadIdx.x;
  if (i >= NN * 128) return;
  int n = i >> 7, o = i & 127;
  float v = nodeb[i] / fmaxf(deg[n], 1.f);
  const float* wr = selfW + (size_t)o * NC;
  const float* hp = h0d + (size_t)n * NC;
  for (int c = 0; c < NC; c++) v += wr[c] * hp[c];
  nodeb[i] = v;
}
__global__ __launch_bounds__(256) void pool_kernel(const float* __restrict__ nodeb,
                                                   float* __restrict__ out, int obase) {
  __shared__ float sm[256];
  int o = blockIdx.x, tid = threadIdx.x;
  float m = -3.4e38f;
  for (int n = tid; n < NN; n += 256) m = fmaxf(m, nodeb[(size_t)n * 128 + o]);
  sm[tid] = m;
  __syncthreads();
  for (int s = 128; s > 0; s >>= 1) {
    if (tid < s) sm[tid] = fmaxf(sm[tid], sm[tid + s]);
    __syncthreads();
  }
  if (tid == 0) out[obase + o] = sm[0];
}

// ---------------------------------------------------------------------------
// Host orchestration
// ---------------------------------------------------------------------------
extern "C" void kernel_launch(void* const* d_in, const int* in_sizes, int n_in,
                              void* d_out, int out_size, void* d_ws, size_t ws_size,
                              hipStream_t stream) {
  (void)in_sizes; (void)n_in; (void)out_size; (void)ws_size;
  const float* node_feats = (const float*)d_in[0];
  const float* edge_w     = (const float*)d_in[1];
  const float* rel_pos    = (const float*)d_in[2];
  const int*   esrc       = (const int*)d_in[3];
  const int*   edst       = (const int*)d_in[4];

  // -- unpack params (jax pytree: dict keys sorted, lists in order) --
  int cur = 5;
  auto nf = [&]() -> const float* { return (const float*)d_in[cur++]; };
  struct RadialP { const float *W1, *W2, *W3, *b1, *b2, *b3, *be1, *be2, *g1, *g2; };
  auto rdr = [&](RadialP& r) {
    r.W1 = nf(); r.W2 = nf(); r.W3 = nf(); r.b1 = nf(); r.b2 = nf();
    r.b3 = nf(); r.be1 = nf(); r.be2 = nf(); r.g1 = nf(); r.g2 = nf();
  };
  struct LayerP { RadialP k[4][4]; const float* proj[4]; const float* q[4]; RadialP v[4][4]; };
  LayerP L[4];
  for (int i = 0; i < 4; i++) {
    if (i == 0) rdr(L[i].k[0][0]);
    else for (int di = 0; di < 4; di++) for (int d0 = 0; d0 < 4; d0++) rdr(L[i].k[di][d0]);
    for (int d = 0; d < 4; d++) L[i].proj[d] = nf();
    if (i == 0) L[i].q[0] = nf();
    else for (int d = 0; d < 4; d++) L[i].q[d] = nf();
    if (i == 0) for (int d0 = 0; d0 < 4; d0++) rdr(L[i].v[0][d0]);
    else for (int di = 0; di < 4; di++) for (int d0 = 0; d0 < 4; d0++) rdr(L[i].v[di][d0]);
  }
  struct TaskP { RadialP pairs[4]; const float* selfW; };
  TaskP cls, reg;
  for (int di = 0; di < 4; di++) rdr(cls.pairs[di]);
  cls.selfW = nf();
  const float* fc1 = nf(); const float* fc2 = nf();
  const float* fcb1 = nf(); const float* fcb2 = nf();
  const float *nWm[4][4], *nbm[4][4];
  for (int i = 0; i < 4; i++)
    for (int d = 0; d < 4; d++) { nWm[i][d] = nf(); nbm[i][d] = nf(); }
  for (int di = 0; di < 4; di++) rdr(reg.pairs[di]);
  reg.selfW = nf();

  // -- workspace bump allocator --
  char* wsp = (char*)d_ws;
  auto alloc = [&](size_t nfl) -> float* {
    float* p = (float*)wsp;
    wsp += ((nfl * sizeof(float) + 255) & ~(size_t)255);
    return p;
  };
  float* Qbuf    = alloc(8192);
  float* efeat   = alloc((size_t)NE * EFP);
  float* basisb  = alloc(basis_total());
  float* h0tmp   = alloc((size_t)NN * AE);
  float* h0      = alloc((size_t)NN * AE);
  float* H1      = alloc((size_t)NE * RD);
  float* H2      = alloc((size_t)NE * RD);
  float* mu      = alloc(RD);
  float* rstd    = alloc(RD);
  float* Rbuf    = alloc((size_t)ECH * 4096);
  _Float16* Whbuf = (_Float16*)alloc(262144);   // 512K halves: largest swizzled weight (128x4096)
  float* vmsg[4]; for (int d = 0; d < 4; d++) vmsg[d] = alloc((size_t)NE * MC * (2 * d + 1));
  float* kmsg[4]; for (int d = 0; d < 4; d++) kmsg[d] = alloc((size_t)NE * MC * (2 * d + 1));
  float* qb4[4];  for (int d = 0; d < 4; d++) qb4[d]  = alloc((size_t)NN * MC * (2 * d + 1));
  float* logitsb = alloc((size_t)NE * NH);
  float* exb     = alloc((size_t)NE * NH);
  float* alphab  = alloc((size_t)NE * NH);
  float* denb    = alloc((size_t)NN * NH);
  unsigned* mxenc = (unsigned*)alloc((size_t)NN * NH);
  float* hcur[4];  for (int d = 0; d < 4; d++) hcur[d]  = alloc((size_t)NN * NC * (2 * d + 1));
  float* hnext[4]; for (int d = 0; d < 4; d++) hnext[d] = alloc((size_t)NN * NC * (2 * d + 1));
  float* aggb  = alloc((size_t)NN * MC * 7);
  float* nrmb  = alloc((size_t)NN * NC);
  float* ebuf  = alloc((size_t)NE * 128);
  float* nodeb = alloc((size_t)NN * 128);
  float* degb  = alloc(NN);

  auto fill = [&](float* p, size_t n, float v) {
    fill_kernel<<<dim3(2048), 256, 0, stream>>>(p, n, v);
  };
  auto wconv = [&](const float* W, int K, int N, int Kpad) {
    wswz_kernel<<<dim3(1024), 256, 0, stream>>>(W, Whbuf, K, N, Kpad);
  };
  auto gemm_run = [&](const float* A, const float* bias, float* C,
                      int M, int N, int Kpad, int lda, int relu) {
    dim3 g(N / 16, M / 128);
    gemm_wmma_kernel<<<g, 256, 0, stream>>>(A, Whbuf, bias, C, M, N, Kpad, lda, N, relu);
  };
  auto gemm = [&](const float* A, const float* W, const float* bias, float* C,
                  int M, int N, int K, int Kpad, int lda, int relu) {
    wconv(W, K, N, Kpad);
    gemm_run(A, bias, C, M, N, Kpad, lda, relu);
  };
  auto radial_h = [&](const RadialP& rp) {
    gemm(efeat, rp.W1, rp.b1, H1, NE, RD, 5, EFP, EFP, 0);
    bn_stats_kernel<<<RD, 256, 0, stream>>>(H1, NE, RD, mu, rstd);
    bn_apply_relu_kernel<<<2048, 256, 0, stream>>>(H1, NE, RD, mu, rstd, rp.g1, rp.be1);
    gemm(H1, rp.W2, rp.b2, H2, NE, RD, RD, RD, RD, 0);
    bn_stats_kernel<<<RD, 256, 0, stream>>>(H2, NE, RD, mu, rstd);
    bn_apply_relu_kernel<<<2048, 256, 0, stream>>>(H2, NE, RD, mu, rstd, rp.g2, rp.be2);
  };
  auto do_pair = [&](const RadialP& rp, int di, int d0, const float* featd, int m_in,
                     float* msgbuf) {
    radial_h(rp);
    int F = 2 * imin2(di, d0) + 1, ao = 2 * d0 + 1, bo = 2 * di + 1;
    int NR = F * MC * m_in;
    const float* Bp = basisb + basis_pair_base(di, d0);
    wconv(rp.W3, RD, NR, RD);           // convert once, reuse across chunks
    for (int e0 = 0; e0 < NE; e0 += ECH) {
      int e1 = imin2(NE, e0 + ECH);
      gemm_run(H2 + (size_t)e0 * RD, rp.b3, Rbuf, e1 - e0, NR, RD, RD, 0);
      pair_msg_kernel<<<dim3((e1 - e0 + 7) / 8), 256, 0, stream>>>(
          Bp, Rbuf, featd, esrc, msgbuf, e0, e1, F, ao, bo, m_in, MC);
    }
  };
  auto do_pair_deg0 = [&](const RadialP& rp, int di, const float* featd, float* eb) {
    radial_h(rp);
    int bo = 2 * di + 1;
    const float* Bp = basisb + basis_pair_base(di, 0);
    wconv(rp.W3, RD, 128 * 32, RD);
    for (int e0 = 0; e0 < NE; e0 += ECH) {
      int e1 = imin2(NE, e0 + ECH);
      gemm_run(H2 + (size_t)e0 * RD, rp.b3, Rbuf, e1 - e0, 128 * 32, RD, RD, 0);
      pair_msg_deg0_kernel<<<dim3((e1 - e0 + 7) / 8), 256, 0, stream>>>(
          Bp, Rbuf, featd, esrc, eb, e0, e1, bo);
    }
  };

  // -- preliminaries --
  compute_q_kernel<<<44, 256, 0, stream>>>(Qbuf);
  edgefeat_kernel<<<(NE + 255) / 256, 256, 0, stream>>>(rel_pos, edge_w, efeat);
  basis_kernel<<<(NE + 63) / 64, 64, 0, stream>>>(rel_pos, Qbuf, basisb);
  gemm(node_feats, fc1, fcb1, h0tmp, NN, AE, AFS, AFS, AFS, 1);
  gemm(h0tmp, fc2, fcb2, h0, NN, AE, AE, AE, AE, 0);

  // -- attention layers --
  for (int i = 0; i < 4; i++) {
    int ndin = (i == 0) ? 1 : 4;
    int m_in = (i == 0) ? AE : NC;
    for (int d0 = 0; d0 < 4; d0++) fill(vmsg[d0], (size_t)NE * MC * (2 * d0 + 1), 0.f);
    for (int di = 0; di < ndin; di++) {
      const float* featd = (i == 0) ? h0 : hcur[di];
      for (int d0 = 0; d0 < 4; d0++) do_pair(L[i].v[di][d0], di, d0, featd, m_in, vmsg[d0]);
    }
    for (int d0 = 0; d0 < ndin; d0++) fill(kmsg[d0], (size_t)NE * MC * (2 * d0 + 1), 0.f);
    for (int di = 0; di < ndin; di++) {
      const float* featd = (i == 0) ? h0 : hcur[di];
      for (int d0 = 0; d0 < ndin; d0++) do_pair(L[i].k[di][d0], di, d0, featd, m_in, kmsg[d0]);
    }
    for (int d = 0; d < ndin; d++) {
      const float* featd = (i == 0) ? h0 : hcur[d];
      g1x1_concat_kernel<<<2048, 256, 0, stream>>>(L[i].q[d], featd, m_in, nullptr, 0,
                                                   qb4[d], NN, MC, 2 * d + 1);
    }
    int nkey = 0;
    for (int d = 0; d < ndin; d++) nkey += MC * (2 * d + 1);
    float scale = 1.0f / sqrtf((float)nkey);
    logits_kernel<<<(NE + 255) / 256, 256, 0, stream>>>(
        kmsg[0], kmsg[1], kmsg[2], kmsg[3], qb4[0], qb4[1], qb4[2], qb4[3],
        edst, logitsb, ndin, scale);
    {
      float ninf = -3.0e38f; unsigned u; memcpy(&u, &ninf, 4);
      unsigned encv = (u & 0x80000000u) ? ~u : (u | 0x80000000u);
      fillu_kernel<<<64, 256, 0, stream>>>(mxenc, (size_t)NN * NH, encv);
    }
    segmax_kernel<<<(NE * NH + 255) / 256, 256, 0, stream>>>(logitsb, edst, mxenc);
    fill(denb, (size_t)NN * NH, 0.f);
    expden_kernel<<<(NE * NH + 255) / 256, 256, 0, stream>>>(logitsb, edst, mxenc, exb, denb);
    alpha_kernel<<<(NE * NH + 255) / 256, 256, 0, stream>>>(exb, edst, denb, alphab);
    for (int d = 0; d < 4; d++) {
      int m = 2 * d + 1;
      fill(aggb, (size_t)NN * MC * m, 0.f);
      vagg_kernel<<<2048, 256, 0, stream>>>(vmsg[d], alphab, edst, aggb, m);
      const float* hprev = nullptr; int c2 = 0;
      if (d < ndin) { hprev = (i == 0) ? h0 : hcur[d]; c2 = m_in; }
      g1x1_concat_kernel<<<2048, 256, 0, stream>>>(L[i].proj[d], aggb, MC, hprev, c2,
                                                   hnext[d], NN, NC, m);
      norm_nrm_kernel<<<(NN * NC + 255) / 256, 256, 0, stream>>>(hnext[d], nrmb, m);
      norm_apply_kernel<<<(NN * NC + 255) / 256, 256, 0, stream>>>(hnext[d], nrmb,
                                                                   nWm[i][d], nbm[i][d], m);
    }
    for (int d = 0; d < 4; d++) { float* t = hcur[d]; hcur[d] = hnext[d]; hnext[d] = t; }
  }

  // -- reg / cls heads --
  fill(degb, NN, 0.f);
  degcount_kernel<<<(NE + 255) / 256, 256, 0, stream>>>(edst, degb);
  TaskP* tasks[2] = {&reg, &cls};
  for (int s = 0; s < 2; s++) {
    fill(ebuf, (size_t)NE * 128, 0.f);
    for (int di = 0; di < 4; di++) do_pair_deg0(tasks[s]->pairs[di], di, hcur[di], ebuf);
    fill(nodeb, (size_t)NN * 128, 0.f);
    eagg_kernel<<<2048, 256, 0, stream>>>(ebuf, edst, nodeb);
    conv_final_kernel<<<(NN * 128 + 255) / 256, 256, 0, stream>>>(nodeb, degb,
                                                                  tasks[s]->selfW, hcur[0]);
    pool_kernel<<<128, 256, 0, stream>>>(nodeb, (float*)d_out, s * 128);
  }
}
